// GConvLSTM_cell_6794638262633
// MI455X (gfx1250) — matrix operand order repaired and checked by reading, compiled, not run on hardware
//
#include <hip/hip_runtime.h>
#include <hip/hip_bf16.h>

typedef __attribute__((ext_vector_type(16))) _Float16 v16h;
typedef __attribute__((ext_vector_type(8)))  float    v8f;

#define CH   64    // per-gate channels
#define CC   128   // combined input channels
#define CO   256   // 4*CH output channels

// ---------------------------------------------------------------- degree init
__global__ void k_init_deg(float* __restrict__ deg, int n) {
    int i = blockIdx.x * blockDim.x + threadIdx.x;
    if (i < n) deg[i] = 1.0f;   // self-loop weight = 1
}

// ------------------------------------------------------- weighted in-degree
__global__ void k_deg_accum(const int* __restrict__ col,
                            const float* __restrict__ w,
                            float* __restrict__ deg, int E, int n) {
    int e = blockIdx.x * blockDim.x + threadIdx.x;
    if (e >= E) return;
    int c = col[e];
    if ((unsigned)c < (unsigned)n)
        __hip_atomic_fetch_add(&deg[c], w[e], __ATOMIC_RELAXED,
                               __HIP_MEMORY_SCOPE_AGENT);
}

// ---------------------------------------------------------------- d^{-1/2}
__global__ void k_dinv(const float* __restrict__ deg,
                       float* __restrict__ dinv, int n) {
    int i = blockIdx.x * blockDim.x + threadIdx.x;
    if (i < n) {
        float d = deg[i];
        dinv[i] = d > 0.0f ? rsqrtf(d) : 0.0f;
    }
}

// --------------------------------------------------- WMMA GEMM xl = [x|h]W^T
// Block = 128 threads = 4 waves. Block handles a 16-row strip of all 256
// output columns; wave w handles columns [64w, 64w+64). K=128 in 4 steps of 32.
__global__ __launch_bounds__(128)
void k_gemm(const float* __restrict__ x, const float* __restrict__ h,
            const float* __restrict__ W, float* __restrict__ xl, int n) {
    const int wave  = threadIdx.x >> 5;
    const int lane  = threadIdx.x & 31;
    const int half  = lane >> 4;     // 0: lanes 0-15, 1: lanes 16-31
    const int l16   = lane & 15;
    const int mBase = blockIdx.x * 16;
    const int nBase = wave * 64;

    int row = mBase + l16;
    if (row >= n) row = n - 1;       // n % 16 == 0 for this problem; safety only

    v8f acc[4] = {v8f{}, v8f{}, v8f{}, v8f{}};

    for (int s = 0; s < 4; ++s) {
        const int kbase = s * 32;
        // ---- A fragment: 16x32 f16. elems 0-7 -> K = kbase+8*half+i,
        //                              elems 8-15 -> K = kbase+8*half+16+i.
        v16h a;
        {
            int k0 = kbase + 8 * half;
            int k1 = k0 + 16;
            const float* s0 = (k0 < CH) ? (x + (size_t)row * CH + k0)
                                        : (h + (size_t)row * CH + (k0 - CH));
            const float* s1 = (k1 < CH) ? (x + (size_t)row * CH + k1)
                                        : (h + (size_t)row * CH + (k1 - CH));
#pragma unroll
            for (int i = 0; i < 8; ++i) a[i]     = (_Float16)s0[i];
#pragma unroll
            for (int i = 0; i < 8; ++i) a[8 + i] = (_Float16)s1[i];
        }
        // ---- B fragments (one per 16-col tile): lane holds column nTile+l16,
        //      elems 0-7 -> K = kbase+8*half+i, elems 8-15 -> K = kbase+8*half+16+i.
#pragma unroll
        for (int t = 0; t < 4; ++t) {
            const int o = nBase + t * 16 + l16;          // output col = W row
            const float* bs = W + (size_t)o * CC + kbase + 8 * half;
            v16h bfrag;
#pragma unroll
            for (int i = 0; i < 16; ++i) bfrag[i] = (_Float16)bs[i < 8 ? i : i + 8];
            acc[t] = __builtin_amdgcn_wmma_f32_16x16x32_f16(
                false, a, false, bfrag, (short)0, acc[t], false, false);
        }
    }
    // ---- store D: element r -> row 8*half + r, column l16
    const int mrow = mBase + 8 * half;
#pragma unroll
    for (int r = 0; r < 8; ++r) {
        const int m = mrow + r;
        if (m < n) {
            float* dst = xl + (size_t)m * CO + nBase + l16;
            dst[0]  = acc[0][r];
            dst[16] = acc[1][r];
            dst[32] = acc[2][r];
            dst[48] = acc[3][r];
        }
    }
}

// ------------------------------------- self-loop message: agg = dinv^2 * xl
__global__ void k_init_agg(const float* __restrict__ dinv,
                           const float* __restrict__ xl,
                           float* __restrict__ agg, int n) {
    size_t i = (size_t)blockIdx.x * blockDim.x + threadIdx.x;  // float4 units
    size_t total = (size_t)n * (CO / 4);
    if (i >= total) return;
    int node = (int)(i >> 6);                 // 64 float4 per node
    float dv = dinv[node];
    float s  = dv * dv;
    const float4 v = ((const float4*)xl)[i];
    float4 r; r.x = s * v.x; r.y = s * v.y; r.z = s * v.z; r.w = s * v.w;
    ((float4*)agg)[i] = r;
}

// -------------------------------------------------- edge gather/scatter-add
// 64 threads per edge; each thread moves one float4 of the 256-wide message.
__global__ __launch_bounds__(256)
void k_edge(const int* __restrict__ row, const int* __restrict__ col,
            const float* __restrict__ w, const float* __restrict__ dinv,
            const float* __restrict__ xl, float* __restrict__ agg,
            int E, int n) {
    int e = blockIdx.x * 4 + (threadIdx.x >> 6);
    if (e >= E) return;
    int r = row[e];
    int c = col[e];
    if ((unsigned)r >= (unsigned)n || (unsigned)c >= (unsigned)n) return;
    float norm = dinv[r] * w[e] * dinv[c];
    int j = threadIdx.x & 63;                             // float4 index 0..63
    const float4 v = ((const float4*)(xl + (size_t)r * CO))[j];
    float* dst = agg + (size_t)c * CO + j * 4;
    __hip_atomic_fetch_add(dst + 0, norm * v.x, __ATOMIC_RELAXED, __HIP_MEMORY_SCOPE_AGENT);
    __hip_atomic_fetch_add(dst + 1, norm * v.y, __ATOMIC_RELAXED, __HIP_MEMORY_SCOPE_AGENT);
    __hip_atomic_fetch_add(dst + 2, norm * v.z, __ATOMIC_RELAXED, __HIP_MEMORY_SCOPE_AGENT);
    __hip_atomic_fetch_add(dst + 3, norm * v.w, __ATOMIC_RELAXED, __HIP_MEMORY_SCOPE_AGENT);
}

// ---------------------------------------------------- fused LSTM gate update
__global__ void k_lstm(const float* __restrict__ agg, const float* __restrict__ b,
                       const float* __restrict__ c_t,
                       float* __restrict__ h_next, float* __restrict__ c_next,
                       int n) {
    int i = blockIdx.x * blockDim.x + threadIdx.x;
    if (i >= n * CH) return;
    int node = i >> 6;
    int j    = i & 63;
    const float* a = agg + (size_t)node * CO;
    float ci = a[j]            + b[j];
    float cf = a[CH + j]       + b[CH + j];
    float co = a[2 * CH + j]   + b[2 * CH + j];
    float cg = a[3 * CH + j]   + b[3 * CH + j];
    float ig = 1.0f / (1.0f + __expf(-ci));
    float fg = 1.0f / (1.0f + __expf(-cf));
    float og = 1.0f / (1.0f + __expf(-co));
    float g  = tanhf(cg);
    float cn = fg * c_t[i] + ig * g;
    float hn = og * tanhf(cn);
    h_next[i] = hn;
    c_next[i] = cn;
}

extern "C" void kernel_launch(void* const* d_in, const int* in_sizes, int n_in,
                              void* d_out, int out_size, void* d_ws, size_t ws_size,
                              hipStream_t stream) {
    const float* x     = (const float*)d_in[0];
    const float* h_t   = (const float*)d_in[1];
    const float* c_t   = (const float*)d_in[2];
    const float* W     = (const float*)d_in[3];
    const float* b     = (const float*)d_in[4];
    const int*   eidx  = (const int*)d_in[5];
    const float* eattr = (const float*)d_in[6];

    const int n = in_sizes[0] / CH;      // 50000
    const int E = in_sizes[6];           // 800000
    const int* erow = eidx;              // edge_index[0] = source
    const int* ecol = eidx + E;          // edge_index[1] = destination

    // workspace layout: deg[n] | dinv[n] | xl[n*256] | agg[n*256]
    float* ws   = (float*)d_ws;
    float* deg  = ws;
    float* dinv = ws + n;
    float* xl   = ws + 2 * (size_t)n;
    float* agg  = xl + (size_t)n * CO;

    float* h_next = (float*)d_out;
    float* c_next = h_next + (size_t)n * CH;

    k_init_deg <<<(n + 255) / 256, 256, 0, stream>>>(deg, n);
    k_deg_accum<<<(E + 255) / 256, 256, 0, stream>>>(ecol, eattr, deg, E, n);
    k_dinv     <<<(n + 255) / 256, 256, 0, stream>>>(deg, dinv, n);
    k_gemm     <<<(n + 15) / 16, 128, 0, stream>>>(x, h_t, W, xl, n);
    {
        size_t total4 = (size_t)n * (CO / 4);
        k_init_agg<<<(unsigned)((total4 + 255) / 256), 256, 0, stream>>>(dinv, xl, agg, n);
    }
    k_edge     <<<(E + 3) / 4, 256, 0, stream>>>(erow, ecol, eattr, dinv, xl, agg, E, n);
    k_lstm     <<<((n * CH) + 255) / 256, 256, 0, stream>>>(agg, b, c_t, h_next, c_next, n);
}
